// RoPEAttention_MA_78348793413910
// MI455X (gfx1250) — compile-verified, matrix operand order checked by reading, not run on hardware
//
#include <hip/hip_runtime.h>
#include <hip/hip_bf16.h>
#include <math.h>

// ---------------------------------------------------------------------------
// RoPE attention for MI455X (gfx1250, wave32, WMMA 16x16x32 f16 -> f32 acc).
// f32->f16 converts, weight pre-pack (frag-linear), WMMA projections with
// RoPE epilogues, split-K flash attention (64-key blocks -> 4 independent
// WMMA accumulation chains, online softmax), combine pass, WMMA O-proj.
// ---------------------------------------------------------------------------

typedef __attribute__((ext_vector_type(16))) _Float16 v16h;
typedef __attribute__((ext_vector_type(8)))  _Float16 v8h;
typedef __attribute__((ext_vector_type(8)))  float    v8f;

#define B_   2
#define SQ_  4096
#define SK_  16448
#define D_   256      // head dim == embed dim (1 head)
#define KV_  64       // kv input dim
#define GW_  64       // sqrt(SQ) for axial rope
#define NS_  4        // split-K factor for flash attention
#define LN1E4_OVER_64 0.14391156831212787f   // ln(10000)/64

static __device__ __forceinline__ v16h cat8(v8h a, v8h b) {
  return __builtin_shufflevector(a, b, 0, 1, 2, 3, 4, 5, 6, 7,
                                       8, 9, 10, 11, 12, 13, 14, 15);
}

// A-frag / B-as-rows frag: lane L holds row (L&15); K elements follow the
// 16-bit 16x32 layout: kb=(L>>4)*8, elems [0..7]=k kb..kb+7, [8..15]=kb+16..kb+23.
static __device__ __forceinline__ v16h frag_rows(const _Float16* p, int ld) {
  int l = threadIdx.x & 31;
  const _Float16* q = p + (size_t)(l & 15) * ld + ((l >> 4) << 3);
  return cat8(*(const v8h*)(q), *(const v8h*)(q + 16));
}

// Frag from pre-packed storage: 512 f16 per 32x16 tile, 16 contiguous per lane.
static __device__ __forceinline__ v16h frag_packed(const _Float16* tile) {
  const _Float16* q = tile + (threadIdx.x & 31) * 16;
  return cat8(*(const v8h*)(q), *(const v8h*)(q + 8));
}

static __device__ __forceinline__ v8f wmma_f16(v16h a, v16h b, v8f c) {
  return __builtin_amdgcn_wmma_f32_16x16x32_f16(false, a, false, b, (short)0, c, false, false);
}

// ---------------------------------------------------------------------------
__global__ void cvt_f16_kernel(const float* __restrict__ s, _Float16* __restrict__ d, int n) {
  for (int i = blockIdx.x * blockDim.x + threadIdx.x; i < n; i += gridDim.x * blockDim.x)
    d[i] = (_Float16)s[i];
}

// Pack W[KxN=256] (fp32 row-major) into frag-linear f16: tile (kt,nt) holds the
// 32x16 block in exact WMMA B-frag per-lane order (16 contiguous f16 per lane).
__global__ __launch_bounds__(128) void pack_w_kernel(
    const float* __restrict__ W, _Float16* __restrict__ Wp, int K) {
  int wid = blockIdx.x * (blockDim.x >> 5) + (threadIdx.x >> 5);
  int tiles = (K / 32) * (D_ / 16);
  if (wid >= tiles) return;
  int kt = wid / (D_ / 16), nt = wid % (D_ / 16);
  int l = threadIdx.x & 31;
  const float* p = W + (size_t)(kt * 32 + ((l >> 4) << 3)) * D_ + nt * 16 + (l & 15);
  v8h s0, s1;
#pragma unroll
  for (int i = 0; i < 8; ++i) {
    s0[i] = (_Float16)p[(size_t)i * D_];
    s1[i] = (_Float16)p[(size_t)(16 + i) * D_];
  }
  _Float16* d = Wp + (size_t)wid * 512 + (size_t)l * 16;
  *(v8h*)(d) = s0;
  *(v8h*)(d + 8) = s1;
}

// ---------------------------------------------------------------------------
// 16x16-tile WMMA GEMM: out = A[MxK] * W[KxN=256] + bias, 1 wave/tile.
// mode 0: Q proj  -> f16 row-major, RoPE, pos = row % SQ, folds 1/sqrt(D)
// mode 1: K proj  -> f16 row-major, RoPE if (row%SK) < SK-excl, pos tiled
// mode 2: V proj  -> f16 TRANSPOSED store vT[(b*D+n)*SK + key]
// mode 3: O proj  -> f32 row-major (d_out)
__global__ __launch_bounds__(128) void proj_wmma_kernel(
    const _Float16* __restrict__ A, const _Float16* __restrict__ Wp,
    const float* __restrict__ bias, void* __restrict__ outp,
    int M, int K, int mode, const int* __restrict__ exclp) {
  int wid = blockIdx.x * (blockDim.x >> 5) + (threadIdx.x >> 5);
  int tilesN = D_ / 16;
  if (wid >= (M / 16) * tilesN) return;
  int tn = wid % tilesN;
  int rb = (wid / tilesN) * 16, cb = tn * 16;
  int l = threadIdx.x & 31;
  int h = l >> 4, nl = l & 15;
  int n = cb + nl;

  v8f c = {};
  for (int k = 0; k < K; k += 32) {
    v16h a = frag_rows(A + (size_t)rb * K + k, K);
    v16h b = frag_packed(Wp + (size_t)((k >> 5) * tilesN + tn) * 512);
    c = wmma_f16(a, b, c);
  }
  float bn = bias[n];
#pragma unroll
  for (int r = 0; r < 8; ++r) c[r] += bn;

  if (mode == 0 || mode == 1) {
    int nkrope = (mode == 1) ? (SK_ - *exclp) : 0;
    float smul = (mode == 0) ? 0.0625f : 1.0f;  // fold 1/sqrt(256) into Q
    int j = n >> 1, par = n & 1;
    float f = __expf(-(float)(j & 63) * LN1E4_OVER_64);  // theta^{-(j&63)/64}
#pragma unroll
    for (int r = 0; r < 8; ++r) {
      int row = rb + r + 8 * h;
      float x = c[r];
      float p = __shfl_xor(x, 1);  // partner column of the rope pair
      int keyrow = (mode == 1) ? (row % SK_) : row;
      bool doro = (mode == 0) || (keyrow < nkrope);
      if (doro) {
        int pos = keyrow % SQ_;
        float t = (j < 64) ? (float)(pos % GW_) : (float)(pos / GW_);
        float sv, cv;
        __sincosf(t * f, &sv, &cv);
        x = (par == 0) ? (x * cv - p * sv) : (p * sv + x * cv);
      }
      c[r] = x * smul;
    }
  }

  if (mode == 2) {                       // transposed f16: contiguous keys
    _Float16* vT = (_Float16*)outp;
    int bidx = rb / SK_;
    int key0 = (rb % SK_) + 8 * h;
    v8h st;
#pragma unroll
    for (int r = 0; r < 8; ++r) st[r] = (_Float16)c[r];
    *(v8h*)(vT + (size_t)(bidx * D_ + n) * SK_ + key0) = st;
  } else if (mode == 3) {
    float* o = (float*)outp;
#pragma unroll
    for (int r = 0; r < 8; ++r) o[(size_t)(rb + r + 8 * h) * D_ + n] = c[r];
  } else {
    _Float16* o = (_Float16*)outp;
#pragma unroll
    for (int r = 0; r < 8; ++r) o[(size_t)(rb + r + 8 * h) * D_ + n] = (_Float16)c[r];
  }
}

// ---------------------------------------------------------------------------
// Split-K flash attention: 1 wave = 16 query rows x one key-range slice,
// processing 64 keys per block (4 independent S WMMA chains).
// Writes unnormalized partials (acc, rowmax m, rowsum l) for the combiner.
__global__ __launch_bounds__(128) void flash_attn_kernel(
    const _Float16* __restrict__ Q, const _Float16* __restrict__ Kh,
    const _Float16* __restrict__ Vt, const unsigned char* __restrict__ mask,
    float* __restrict__ accP, float* __restrict__ mP, float* __restrict__ lP) {
  __shared__ _Float16 qlds[4][16][D_];   // per-wave staged Q tile (8 KB each)
  __shared__ _Float16 plds[4][16][64];   // per-wave P transpose buffer (2 KB)
  int w = threadIdx.x >> 5;
  int wid = blockIdx.x * (blockDim.x >> 5) + w;
  int ntile = B_ * (SQ_ / 16);
  if (wid >= ntile * NS_) return;
  int tile = wid / NS_, s = wid % NS_;
  int b = tile / (SQ_ / 16);
  int qb = (tile % (SQ_ / 16)) * 16;
  int l = threadIdx.x & 31;
  int h = l >> 4, nl = l & 15;

  // key-block range for this split (64-key blocks; SK = 64*257)
  int blocksTot = SK_ / 64;                        // 257
  int per = (blocksTot + NS_ - 1) / NS_;           // 65
  int kb0 = s * per * 64;
  int kbend = min(SK_, (s * per + per) * 64);

  // Stage the contiguous 16x256 Q tile into LDS (wave-local, DS is in-order).
  {
    const v8h* src = (const v8h*)(Q + (size_t)(b * SQ_ + qb) * D_);
    v8h* dst = (v8h*)&qlds[w][0][0];
    for (int i = l; i < 16 * D_ / 8; i += 32) dst[i] = src[i];
  }

  v8f zero = {};
  v8f acc[16];
#pragma unroll
  for (int t = 0; t < 16; ++t) acc[t] = zero;
  float mi[8], li[8];
#pragma unroll
  for (int r = 0; r < 8; ++r) { mi[r] = -3.0e38f; li[r] = 0.0f; }

  const unsigned char* mb = mask + (size_t)b * SK_;

  for (int kb = kb0; kb < kbend; kb += 64) {
    // Prefetch K rows 128 keys ahead (global_prefetch_b8; speculative-safe).
    int pfk = kb + 128 + l;
    if (pfk < SK_) __builtin_prefetch(Kh + (size_t)(b * SK_ + pfk) * D_, 0, 1);
    if (pfk + 32 < SK_) __builtin_prefetch(Kh + (size_t)(b * SK_ + pfk + 32) * D_, 0, 1);

    // S = Q * K^T for 64 keys (four 16x16 C-frags = 4 independent chains).
    v8f s0 = zero, s1 = zero, s2 = zero, s3 = zero;
#pragma unroll
    for (int j = 0; j < 8; ++j) {
      v16h a  = frag_rows(&qlds[w][0][0] + 32 * j, D_);
      const _Float16* kp = Kh + (size_t)(b * SK_ + kb) * D_ + 32 * j;
      v16h b0 = frag_rows(kp,             D_);
      v16h b1 = frag_rows(kp + 16 * D_,   D_);
      v16h b2 = frag_rows(kp + 32 * D_,   D_);
      v16h b3 = frag_rows(kp + 48 * D_,   D_);
      s0 = wmma_f16(a, b0, s0);
      s1 = wmma_f16(a, b1, s1);
      s2 = wmma_f16(a, b2, s2);
      s3 = wmma_f16(a, b3, s3);
    }
    bool m0 = mb[kb + nl]      != 0;     // mask is per-key -> lane-local
    bool m1 = mb[kb + 16 + nl] != 0;
    bool m2 = mb[kb + 32 + nl] != 0;
    bool m3 = mb[kb + 48 + nl] != 0;

    // Online softmax per row r (row m = r + 8h lives across the 16-lane half).
#pragma unroll
    for (int r = 0; r < 8; ++r) {
      float a0 = m0 ? s0[r] : -3.0e38f;  // 1/sqrt(D) already folded into Q
      float a1 = m1 ? s1[r] : -3.0e38f;
      float a2 = m2 ? s2[r] : -3.0e38f;
      float a3 = m3 ? s3[r] : -3.0e38f;
      float bm = fmaxf(fmaxf(a0, a1), fmaxf(a2, a3));
      bm = fmaxf(bm, __shfl_xor(bm, 1));
      bm = fmaxf(bm, __shfl_xor(bm, 2));
      bm = fmaxf(bm, __shfl_xor(bm, 4));
      bm = fmaxf(bm, __shfl_xor(bm, 8));
      float mn = fmaxf(mi[r], bm);
      float sc = __expf(mi[r] - mn);
      float p0 = m0 ? __expf(a0 - mn) : 0.0f;
      float p1 = m1 ? __expf(a1 - mn) : 0.0f;
      float p2 = m2 ? __expf(a2 - mn) : 0.0f;
      float p3 = m3 ? __expf(a3 - mn) : 0.0f;
      float rs = (p0 + p1) + (p2 + p3);
      rs += __shfl_xor(rs, 1);
      rs += __shfl_xor(rs, 2);
      rs += __shfl_xor(rs, 4);
      rs += __shfl_xor(rs, 8);
      li[r] = li[r] * sc + rs;
      mi[r] = mn;
#pragma unroll
      for (int t = 0; t < 16; ++t) acc[t][r] *= sc;
      plds[w][r + 8 * h][nl]      = (_Float16)p0;   // transpose P via LDS
      plds[w][r + 8 * h][nl + 16] = (_Float16)p1;
      plds[w][r + 8 * h][nl + 32] = (_Float16)p2;
      plds[w][r + 8 * h][nl + 48] = (_Float16)p3;
    }

    // acc += P(16x64) * V(64x256) as two K=32 passes; one P frag reused over
    // all 16 N-tiles per pass, Vt rows give contiguous keys per dim.
    v16h pa0 = frag_rows(&plds[w][0][0], 64);
#pragma unroll
    for (int t = 0; t < 16; ++t) {
      v16h bv = frag_rows(Vt + (size_t)(b * D_ + t * 16) * SK_ + kb, SK_);
      acc[t] = wmma_f16(pa0, bv, acc[t]);
    }
    v16h pa1 = frag_rows(&plds[w][0][0] + 32, 64);
#pragma unroll
    for (int t = 0; t < 16; ++t) {
      v16h bv = frag_rows(Vt + (size_t)(b * D_ + t * 16) * SK_ + kb + 32, SK_);
      acc[t] = wmma_f16(pa1, bv, acc[t]);
    }
  }

  // Store unnormalized partials for the combiner.
  size_t pbase = (size_t)wid * 16;
#pragma unroll
  for (int t = 0; t < 16; ++t)
#pragma unroll
    for (int r = 0; r < 8; ++r)
      accP[(pbase + r + 8 * h) * D_ + t * 16 + nl] = acc[t][r];
  if (nl == 0) {
#pragma unroll
    for (int r = 0; r < 8; ++r) {
      mP[pbase + r + 8 * h] = mi[r];
      lP[pbase + r + 8 * h] = li[r];
    }
  }
}

// Merge NS_ split-K partials per query row, normalize, emit f16 for O-proj.
__global__ void flash_combine_kernel(const float* __restrict__ accP,
                                     const float* __restrict__ mP,
                                     const float* __restrict__ lP,
                                     _Float16* __restrict__ O) {
  int idx = blockIdx.x * blockDim.x + threadIdx.x;
  if (idx >= B_ * SQ_ * D_) return;
  int d = idx & (D_ - 1);
  int row = idx >> 8;
  int tile = row >> 4, rr = row & 15;
  int base = tile * NS_;
  float gm = -3.0e38f;
#pragma unroll
  for (int s = 0; s < NS_; ++s) gm = fmaxf(gm, mP[(base + s) * 16 + rr]);
  float L = 0.0f, o = 0.0f;
#pragma unroll
  for (int s = 0; s < NS_; ++s) {
    float wgt = __expf(mP[(base + s) * 16 + rr] - gm);
    L += lP[(base + s) * 16 + rr] * wgt;
    o += accP[((size_t)(base + s) * 16 + rr) * D_ + d] * wgt;
  }
  O[idx] = (_Float16)(o / L);
}

// ---------------------------------------------------------------------------
extern "C" void kernel_launch(void* const* d_in, const int* in_sizes, int n_in,
                              void* d_out, int out_size, void* d_ws, size_t ws_size,
                              hipStream_t stream) {
  (void)in_sizes; (void)n_in; (void)out_size; (void)ws_size;
  const float* q  = (const float*)d_in[0];
  const float* k  = (const float*)d_in[1];
  const float* v  = (const float*)d_in[2];
  const unsigned char* mask = (const unsigned char*)d_in[3];  // jax bool = 1 byte
  const float* wq = (const float*)d_in[4];
  const float* bq = (const float*)d_in[5];
  const float* wk = (const float*)d_in[6];
  const float* bk = (const float*)d_in[7];
  const float* wv = (const float*)d_in[8];
  const float* bv = (const float*)d_in[9];
  const float* wo = (const float*)d_in[10];
  const float* bo = (const float*)d_in[11];
  const int* excl = (const int*)d_in[12];

  char* ws = (char*)d_ws;
  size_t off = 0;
  auto alloc = [&](size_t bytes) { void* p = ws + off; off += (bytes + 255) & ~(size_t)255; return p; };
  _Float16* q16  = (_Float16*)alloc((size_t)B_ * SQ_ * D_  * 2);
  _Float16* k16  = (_Float16*)alloc((size_t)B_ * SK_ * KV_ * 2);
  _Float16* v16  = (_Float16*)alloc((size_t)B_ * SK_ * KV_ * 2);
  _Float16* wqp  = (_Float16*)alloc((size_t)D_  * D_ * 2);
  _Float16* wkp  = (_Float16*)alloc((size_t)KV_ * D_ * 2);
  _Float16* wvp  = (_Float16*)alloc((size_t)KV_ * D_ * 2);
  _Float16* wop  = (_Float16*)alloc((size_t)D_  * D_ * 2);
  _Float16* qh   = (_Float16*)alloc((size_t)B_ * SQ_ * D_ * 2);
  _Float16* kh   = (_Float16*)alloc((size_t)B_ * SK_ * D_ * 2);
  _Float16* vT   = (_Float16*)alloc((size_t)B_ * D_ * SK_ * 2);
  _Float16* at16 = (_Float16*)alloc((size_t)B_ * SQ_ * D_ * 2);
  int ntile = B_ * (SQ_ / 16);                       // 512
  float* accP = (float*)alloc((size_t)ntile * NS_ * 16 * D_ * 4);  // 33.5 MB
  float* mP   = (float*)alloc((size_t)ntile * NS_ * 16 * 4);
  float* lP   = (float*)alloc((size_t)ntile * NS_ * 16 * 4);

  cvt_f16_kernel<<<1024, 256, 0, stream>>>(q, q16, B_ * SQ_ * D_);
  cvt_f16_kernel<<<1024, 256, 0, stream>>>(k, k16, B_ * SK_ * KV_);
  cvt_f16_kernel<<<1024, 256, 0, stream>>>(v, v16, B_ * SK_ * KV_);
  pack_w_kernel<<<32, 128, 0, stream>>>(wq, wqp, D_);
  pack_w_kernel<<<8,  128, 0, stream>>>(wk, wkp, KV_);
  pack_w_kernel<<<8,  128, 0, stream>>>(wv, wvp, KV_);
  pack_w_kernel<<<32, 128, 0, stream>>>(wo, wop, D_);

  int tilesQ = (B_ * SQ_ / 16) * (D_ / 16);   // 8192
  int tilesK = (B_ * SK_ / 16) * (D_ / 16);   // 32896
  proj_wmma_kernel<<<(tilesQ + 3) / 4, 128, 0, stream>>>(q16, wqp, bq, qh, B_ * SQ_, D_,  0, nullptr);
  proj_wmma_kernel<<<(tilesK + 3) / 4, 128, 0, stream>>>(k16, wkp, bk, kh, B_ * SK_, KV_, 1, excl);
  proj_wmma_kernel<<<(tilesK + 3) / 4, 128, 0, stream>>>(v16, wvp, bv, vT, B_ * SK_, KV_, 2, nullptr);

  flash_attn_kernel<<<(ntile * NS_ + 3) / 4, 128, 0, stream>>>(qh, kh, vT, mask, accP, mP, lP);
  flash_combine_kernel<<<(B_ * SQ_ * D_ + 255) / 256, 256, 0, stream>>>(accP, mP, lP, at16);

  proj_wmma_kernel<<<(tilesQ + 3) / 4, 128, 0, stream>>>(at16, wop, bo, d_out, B_ * SQ_, D_, 3, nullptr);
}
